// AffineCouplingTransform_28260884807682
// MI455X (gfx1250) — matrix operand order repaired and checked by reading, not compiled, not run on hardware
//
#include <hip/hip_runtime.h>
#include <hip/hip_bf16.h>

// ---------- types for WMMA ----------
typedef __attribute__((ext_vector_type(16))) __bf16 bf16x16;
typedef __attribute__((ext_vector_type(8)))  float  f32x8;

// round-to-nearest-even fp32 -> bf16
__device__ __forceinline__ unsigned short f2bf(float f) {
    unsigned int u = __float_as_uint(f);
    return (unsigned short)((u + 0x7FFFu + ((u >> 16) & 1u)) >> 16);
}

__device__ __forceinline__ unsigned int pack2bf(float a, float b) {
    return (unsigned)f2bf(a) | ((unsigned)f2bf(b) << 16);
}

// two 16B LDS reads -> one 16x32 (or 32x16) bf16 fragment
__device__ __forceinline__ bf16x16 make_frag(const unsigned int* p0, const unsigned int* p1) {
    union { uint4 q[2]; bf16x16 v; } u;
    u.q[0] = *(const uint4*)p0;
    u.q[1] = *(const uint4*)p1;
    return u.v;
}

// CDNA5 async copy: 16B global -> LDS, tracked by ASYNCcnt (no VGPR staging)
__device__ __forceinline__ void async_b128(unsigned lds_byte_off, const void* gaddr) {
    asm volatile("global_load_async_to_lds_b128 %0, %1, off"
                 :: "v"(lds_byte_off), "v"(gaddr)
                 : "memory");
}

__device__ __forceinline__ void wait_async0() {
#if defined(__has_builtin)
#if __has_builtin(__builtin_amdgcn_s_wait_asynccnt)
    __builtin_amdgcn_s_wait_asynccnt(0);
    return;
#endif
#endif
    asm volatile("s_wait_asynccnt 0x0" ::: "memory");
}

#define BM 128
#define BN 128
#define BK 32
#define LDW 20   // uints per LDS row: 16 data + 4 pad -> 80B (16B aligned, conflict-free)

// ----------------------------------------------------------------------------
// Tiled bf16 WMMA GEMM:  C[BxN] = act(A[BxK] @ W[KxN] + bias)
// A is bf16 [B][K]; W supplied PRE-TRANSPOSED as Wt[N][K] bf16 so both tiles
// live in LDS as K-contiguous rows, read with k = 16*lhi + s (packed K-pairs per row).
// Block: 256 threads = 8 waves (2x4), each wave 64x32 via 4x2 wmma tiles.
// Double-buffered LDS filled by GLOBAL_LOAD_ASYNC_TO_LDS_B128: tile k+1
// streams into LDS while the 8 WMMAs of tile k execute; no staged VGPRs.
// ----------------------------------------------------------------------------
template <bool RELU_BF16>
__global__ __launch_bounds__(256) void gemm_bf16_kernel(
    const unsigned short* __restrict__ A,        // B x K bf16
    const unsigned short* __restrict__ Wt,       // N x K bf16 (pre-transposed)
    const float* __restrict__ bias,              // N
    void* __restrict__ Cptr,                     // bf16 [B][N] or fp32 [B][N]
    int K, int N)
{
    __shared__ __align__(16) unsigned int As[2][BM][LDW];
    __shared__ __align__(16) unsigned int Bs[2][BN][LDW];

    const int tid      = threadIdx.x;
    const int rowBlock = blockIdx.y * BM;
    const int colBlock = blockIdx.x * BN;

    const int w     = tid >> 5;
    const int lane  = tid & 31;
    const int lm    = lane & 15;
    const int lhi   = lane >> 4;
    const int waveM = (w >> 2) * 64;
    const int waveN = (w & 3) * 32;

    const int r   = tid >> 1;   // 0..127 : tile row (A) / tile col (B)
    const int seg = tid & 1;    // which 16-K half

    // LDS byte offsets of this thread's staging slots (low 32 bits of flat addr)
    const unsigned asBase = (unsigned)(size_t)(&As[0][0][0]);
    const unsigned bsBase = (unsigned)(size_t)(&Bs[0][0][0]);
    const unsigned slotA0 = asBase + (unsigned)((r * LDW + seg * 8) * 4);
    const unsigned slotB0 = bsBase + (unsigned)((r * LDW + seg * 8) * 4);
    const unsigned bufStrideA = (unsigned)(BM * LDW * 4);
    const unsigned bufStrideB = (unsigned)(BN * LDW * 4);

    f32x8 acc[4][2] = {};

    const int nsteps = K / BK;

    auto stage_async = [&](int buf, int k0) {
        const unsigned short* ga = A  + (size_t)(rowBlock + r) * K + k0 + seg * 16;
        const unsigned short* gb = Wt + (size_t)(colBlock + r) * K + k0 + seg * 16;
        const unsigned la = slotA0 + (unsigned)buf * bufStrideA;
        const unsigned lb = slotB0 + (unsigned)buf * bufStrideB;
        async_b128(la,      ga);
        async_b128(la + 16, ga + 8);
        async_b128(lb,      gb);
        async_b128(lb + 16, gb + 8);
    };

    // ---- prologue: stream tile 0 into buffer 0 ----
    stage_async(0, 0);
    wait_async0();
    __syncthreads();

    for (int step = 0; step < nsteps; ++step) {
        const int  cur      = step & 1;
        const bool has_next = (step + 1) < nsteps;

        // stream tile k+1 into the other buffer under the WMMAs
        if (has_next) stage_async(cur ^ 1, (step + 1) * BK);

        // fragments from LDS (2x ds_load_b128 each, k = 16*lhi + s in A and B)
        bf16x16 a[4], b[2];
        #pragma unroll
        for (int nt = 0; nt < 2; ++nt) {
            const unsigned int* col = Bs[cur][waveN + nt * 16 + lm];
            b[nt] = make_frag(col + lhi * 8, col + lhi * 8 + 4);
        }
        #pragma unroll
        for (int mt = 0; mt < 4; ++mt) {
            const unsigned int* row = As[cur][waveM + mt * 16 + lm];
            a[mt] = make_frag(row + lhi * 8, row + lhi * 8 + 4);
        }

        #pragma unroll
        for (int mt = 0; mt < 4; ++mt)
            #pragma unroll
            for (int nt = 0; nt < 2; ++nt)
                acc[mt][nt] = __builtin_amdgcn_wmma_f32_16x16x32_bf16(
                    false, a[mt], false, b[nt],
                    (short)0, acc[mt][nt], false, false);

        if (has_next) wait_async0();   // own async copies done ...
        __syncthreads();               // ... all waves' copies visible
    }

    // ---- epilogue: bias (+relu->bf16 | fp32) ----
    #pragma unroll
    for (int nt = 0; nt < 2; ++nt) {
        const int col = colBlock + waveN + nt * 16 + lm;
        const float bv = bias[col];
        #pragma unroll
        for (int mt = 0; mt < 4; ++mt) {
            #pragma unroll
            for (int rr = 0; rr < 8; ++rr) {
                const int row = rowBlock + waveM + mt * 16 + lhi * 8 + rr;
                float v = acc[mt][nt][rr] + bv;
                if (RELU_BF16) {
                    v = v > 0.0f ? v : 0.0f;
                    ((unsigned short*)Cptr)[(size_t)row * N + col] = f2bf(v);
                } else {
                    ((float*)Cptr)[(size_t)row * N + col] = v;
                }
            }
        }
    }
}

// ---------- fp32 W[K][N] -> bf16 Wt[N][K] (transpose + convert) ----------
__global__ void convert_transpose_bf16_kernel(const float* __restrict__ src,
                                              unsigned short* __restrict__ dst,
                                              int K, int N) {
    int i = blockIdx.x * blockDim.x + threadIdx.x;
    if (i < K * N) {
        int k = i / N, n = i % N;
        dst[(size_t)n * K + k] = f2bf(src[i]);
    }
}

// ---------- x[:, :256] fp32 -> xb[B][256] bf16 (vectorized) ----------
__global__ __launch_bounds__(256) void convert_x_bf16_kernel(
    const float* __restrict__ x,          // B x 512
    unsigned short* __restrict__ xb)      // B x 256
{
    const int i   = blockIdx.x * blockDim.x + threadIdx.x;  // one per 4 elems
    const int row = i >> 6;          // 256/4 = 64 groups per row
    const int col = (i & 63) * 4;
    float4 f = *(const float4*)(x + (size_t)row * 512 + col);
    uint2 q;
    q.x = pack2bf(f.x, f.y);
    q.y = pack2bf(f.z, f.w);
    *(uint2*)(xb + (size_t)row * 256 + col) = q;
}

// ---------- coupling epilogue: one 256-thread block per row ----------
__global__ __launch_bounds__(256) void coupling_epilogue_kernel(
    const float* __restrict__ x,       // B x 512
    const float* __restrict__ params,  // B x 512 (shift | unconstrained scale)
    float* __restrict__ out,           // B x 512
    float* __restrict__ logabsdet)     // B
{
    const int row = blockIdx.x;
    const int j   = threadIdx.x;   // 0..255
    const float* xr   = x      + (size_t)row * 512;
    const float* pr   = params + (size_t)row * 512;
    float*       orow = out    + (size_t)row * 512;

    // identity half
    orow[j] = xr[j];

    // transform half
    const float shift = pr[j];
    const float us    = pr[256 + j];
    const float s     = 1.0f / (1.0f + expf(-(us + 2.0f))) + 0.001f;
    orow[256 + j] = xr[256 + j] * s + shift;

    // row reduction of log(scale): wave32 shuffle + LDS
    float ls = logf(s);
    #pragma unroll
    for (int o = 16; o > 0; o >>= 1) ls += __shfl_down(ls, o);
    __shared__ float part[8];
    if ((threadIdx.x & 31) == 0) part[threadIdx.x >> 5] = ls;
    __syncthreads();
    if (threadIdx.x == 0) {
        float t = 0.0f;
        #pragma unroll
        for (int i = 0; i < 8; ++i) t += part[i];
        logabsdet[row] = t;
    }
}

extern "C" void kernel_launch(void* const* d_in, const int* in_sizes, int n_in,
                              void* d_out, int out_size, void* d_ws, size_t ws_size,
                              hipStream_t stream) {
    (void)in_sizes; (void)n_in; (void)out_size; (void)ws_size;
    const float* x  = (const float*)d_in[0];
    const float* W1 = (const float*)d_in[1];
    const float* b1 = (const float*)d_in[2];
    const float* W2 = (const float*)d_in[3];
    const float* b2 = (const float*)d_in[4];
    const float* W3 = (const float*)d_in[5];
    const float* b3 = (const float*)d_in[6];
    const float* W4 = (const float*)d_in[7];
    const float* b4 = (const float*)d_in[8];

    const int B = 65536, H = 1024, DID = 256;

    // workspace layout (weights stored TRANSPOSED: Wt[N][K] bf16)
    char* ws = (char*)d_ws;
    unsigned short* Wt1 = (unsigned short*)ws;                 // 1024 x 256
    unsigned short* Wt2 = Wt1 + DID * H;                       // 1024 x 1024
    unsigned short* Wt3 = Wt2 + H * H;                         // 1024 x 1024
    unsigned short* Wt4 = Wt3 + H * H;                         // 512 x 1024
    unsigned short* xb  = (unsigned short*)(ws + (8ull  << 20));   // B x 256 bf16 (32MB)
    unsigned short* hA  = (unsigned short*)(ws + (48ull << 20));   // B x 1024 bf16 (128MB)
    unsigned short* hB  = (unsigned short*)(ws + (176ull << 20));  // B x 1024 bf16 (128MB)
    float* params = (float*)hB;  // B x 512 fp32 aliases hB (free after layer 3)

    // 1) convert + transpose weights; convert x identity half to bf16
    convert_transpose_bf16_kernel<<<(DID * H + 255) / 256, 256, 0, stream>>>(W1, Wt1, DID, H);
    convert_transpose_bf16_kernel<<<(H * H + 255) / 256, 256, 0, stream>>>(W2, Wt2, H, H);
    convert_transpose_bf16_kernel<<<(H * H + 255) / 256, 256, 0, stream>>>(W3, Wt3, H, H);
    convert_transpose_bf16_kernel<<<(H * 512 + 255) / 256, 256, 0, stream>>>(W4, Wt4, H, 512);
    convert_x_bf16_kernel<<<(B * DID / 4) / 256, 256, 0, stream>>>(x, xb);

    // 2) MLP layers (all pure-bf16 WMMA path, async-LDS pipelined)
    dim3 blk(256);
    gemm_bf16_kernel<true><<<dim3(H / BN, B / BM), blk, 0, stream>>>(
        xb, Wt1, b1, hA, DID, H);
    gemm_bf16_kernel<true><<<dim3(H / BN, B / BM), blk, 0, stream>>>(
        hA, Wt2, b2, hB, H, H);
    gemm_bf16_kernel<true><<<dim3(H / BN, B / BM), blk, 0, stream>>>(
        hB, Wt3, b3, hA, H, H);
    gemm_bf16_kernel<false><<<dim3(512 / BN, B / BM), blk, 0, stream>>>(
        hA, Wt4, b4, params, H, 512);

    // 3) coupling transform + logabsdet
    float* out = (float*)d_out;
    float* lad = out + (size_t)B * 512;
    coupling_epilogue_kernel<<<B, 256, 0, stream>>>(x, params, out, lad);
}
